// BitModel_89799176225622
// MI455X (gfx1250) — matrix operand order repaired and checked by reading, not compiled, not run on hardware
//
#include <hip/hip_runtime.h>
#include <hip/hip_bf16.h>
#include <stdint.h>

typedef __attribute__((ext_vector_type(16))) __bf16 v16bf;
typedef __attribute__((ext_vector_type(8)))  __bf16 v8bf;
typedef __attribute__((ext_vector_type(8)))  float  v8f;

#define BK 32

__device__ __forceinline__ __bf16 f2bf(float f) {
  // round-to-nearest-even f32 -> bf16, branch free
  unsigned u = __builtin_bit_cast(unsigned, f);
  unsigned r = (u + 0x7FFFu + ((u >> 16) & 1u)) >> 16;
  unsigned short s = (unsigned short)r;
  return __builtin_bit_cast(__bf16, s);
}

__device__ __forceinline__ __bf16 ternbf(float w) {
  // ternary weight: 0 if |w| < 0.33 else sign(w); exact bf16 bit patterns
  unsigned short s = (fabsf(w) < 0.33f) ? (unsigned short)0x0000
                   : ((w < 0.0f) ? (unsigned short)0xBF80 : (unsigned short)0x3F80);
  return __builtin_bit_cast(__bf16, s);
}

__device__ __forceinline__ float sigm(float x) { return 1.0f / (1.0f + __expf(-x)); }
__device__ __forceinline__ float siluf(float x) { return x * sigm(x); }

// MODE 0: A = f32 (x), NB=3 (f,c,g) -> gh = sig(g)*(1-sig(f))*silu(c), out bf16
// MODE 1: A = bf16,    NB=1        -> acc + bias, out bf16
// MODE 2: A = bf16,    NB=2 (g,u)  -> sig(g)*silu(u), out bf16
// MODE 3: A = bf16,    NB=1        -> acc + bias, out f32
template <int MODE>
__global__ __launch_bounds__(256, 2)
void bit_gemm(const void* __restrict__ Ap,
              const float* __restrict__ B0,
              const float* __restrict__ B1,
              const float* __restrict__ B2,
              const float* __restrict__ bias0,
              const float* __restrict__ bias1,
              const float* __restrict__ bias2,
              void* __restrict__ Outp,
              int M, int N, int K)
{
  constexpr int  NB  = (MODE == 0) ? 3 : (MODE == 2) ? 2 : 1;
  constexpr bool ABF = (MODE != 0);
  // register-tile geometry: ds_b128 per WMMA = (2*FM + 2*NB*FN) / (NB*FM*FN)
  constexpr int  FM  = (MODE == 0) ? 2 : 4;   // 16-row A frags per wave
  constexpr int  FN  = (MODE == 3) ? 4 : 2;   // 16-col B frags per wave
  constexpr int  WGM = (MODE == 0) ? 4 : 2;   // wave grid (8 waves total)
  constexpr int  WGN = (MODE == 0) ? 2 : 4;
  constexpr int  BM  = WGM * FM * 16;         // 128 in all modes
  constexpr int  BN  = WGN * FN * 16;         // 64 / 128 / 128 / 256
  (void)M;

  __shared__ __align__(64) __bf16 lsA[2][BM][BK];          // A tile, row-major [m][k]
  __shared__ __align__(64) __bf16 lsB[2][NB][BN][BK];      // B tile transposed [n][k]

  const int t    = threadIdx.x;
  const int lane = t & 31;
  const int wave = t >> 5;          // 8 waves (wave32)
  const int wm   = wave / WGN;
  const int wn   = wave % WGN;
  const int m0   = blockIdx.y * BM;
  const int n0   = blockIdx.x * BN;

  const float* Bmat[3] = {B0, B1, B2};

  v8f acc[FM][FN][NB];
#pragma unroll
  for (int fm = 0; fm < FM; ++fm)
#pragma unroll
    for (int fn = 0; fn < FN; ++fn)
#pragma unroll
      for (int b = 0; b < NB; ++b)
        acc[fm][fn][b] = (v8f){0.f, 0.f, 0.f, 0.f, 0.f, 0.f, 0.f, 0.f};

  const int KT = K / BK;

  auto loadA = [&](int buf, int kt) {
    const int r = t >> 1;           // 0..127 (BM==128)
    const int h = t & 1;            // 16-element half of a 32-wide K row
    if constexpr (ABF) {
      // raw bf16 bytes: CDNA5 async copy global -> LDS (ASYNCcnt tracked)
      const __bf16* g = (const __bf16*)Ap + (size_t)(m0 + r) * K + (size_t)kt * BK + h * 16;
      unsigned ldsa = (unsigned)(size_t)&lsA[buf][r][h * 16];
      asm volatile("global_load_async_to_lds_b128 %0, %1, off"
                   :: "v"(ldsa), "v"(g) : "memory");
      asm volatile("global_load_async_to_lds_b128 %0, %1, off offset:16"
                   :: "v"(ldsa), "v"(g) : "memory");
    } else {
      // f32 activations -> bf16 convert on the fly
      const float* g = (const float*)Ap + (size_t)(m0 + r) * K + (size_t)kt * BK + h * 16;
      float4 f0 = ((const float4*)g)[0];
      float4 f1 = ((const float4*)g)[1];
      float4 f2 = ((const float4*)g)[2];
      float4 f3 = ((const float4*)g)[3];
      __bf16 tmp[16];
      tmp[0]=f2bf(f0.x);  tmp[1]=f2bf(f0.y);  tmp[2]=f2bf(f0.z);  tmp[3]=f2bf(f0.w);
      tmp[4]=f2bf(f1.x);  tmp[5]=f2bf(f1.y);  tmp[6]=f2bf(f1.z);  tmp[7]=f2bf(f1.w);
      tmp[8]=f2bf(f2.x);  tmp[9]=f2bf(f2.y);  tmp[10]=f2bf(f2.z); tmp[11]=f2bf(f2.w);
      tmp[12]=f2bf(f3.x); tmp[13]=f2bf(f3.y); tmp[14]=f2bf(f3.z); tmp[15]=f2bf(f3.w);
      v8bf* d = (v8bf*)&lsA[buf][r][h * 16];
      d[0] = *(const v8bf*)&tmp[0];
      d[1] = *(const v8bf*)&tmp[8];
    }
  };

  auto loadB = [&](int buf, int kt) {
    // NB tiles of BN x 32 f32, ternarized; 8-float chunks, BN/64 chunks/thread/tile
    constexpr int C = BN / 64;
#pragma unroll
    for (int b = 0; b < NB; ++b) {
#pragma unroll
      for (int c = 0; c < C; ++c) {
        const int id = t + c * 256;
        const int r  = id >> 2;     // output column n within tile
        const int q  = id & 3;      // 8-float K chunk
        const float* g = Bmat[b] + (size_t)(n0 + r) * K + (size_t)kt * BK + q * 8;
        float4 f0 = ((const float4*)g)[0];
        float4 f1 = ((const float4*)g)[1];
        __bf16 tb[8];
        tb[0]=ternbf(f0.x); tb[1]=ternbf(f0.y); tb[2]=ternbf(f0.z); tb[3]=ternbf(f0.w);
        tb[4]=ternbf(f1.x); tb[5]=ternbf(f1.y); tb[6]=ternbf(f1.z); tb[7]=ternbf(f1.w);
        *(v8bf*)&lsB[buf][b][r][q * 8] = *(const v8bf*)&tb[0];
      }
    }
  };

  // bf16 WMMA per-lane fragment order chosen here (contiguous k, same for A and B):
  //  A: row m = lane%16; lanes 0-15 hold K 0-15, lanes 16-31 K 16-31 (contiguous)
  //  B: col n = lane%16; lanes 0-15 hold K 0-15, lanes 16-31 K 16-31 (contiguous)
  const int kbA = (lane >> 4) * 16;
  const int rA0 = wm * (FM * 16) + (lane & 15);
  const int kbB = (lane >> 4) * 16;
  const int rB0 = wn * (FN * 16) + (lane & 15);

  loadA(0, 0);
  loadB(0, 0);
  int buf = 0;
  for (int kt = 0; kt < KT; ++kt) {
    if constexpr (ABF)
      asm volatile("s_wait_asynccnt 0" ::: "memory");   // my async LDS writes landed
    __syncthreads();                                    // publish tile `buf`
    if (kt + 1 < KT) { loadA(buf ^ 1, kt + 1); loadB(buf ^ 1, kt + 1); }

    v16bf afr[FM];
#pragma unroll
    for (int fm = 0; fm < FM; ++fm) {
      const __bf16* p = &lsA[buf][rA0 + fm * 16][0];
      v8bf lo = *(const v8bf*)(p + kbA);
      v8bf hi = *(const v8bf*)(p + 8 + kbA);
      afr[fm] = __builtin_shufflevector(lo, hi, 0,1,2,3,4,5,6,7,8,9,10,11,12,13,14,15);
    }
#pragma unroll
    for (int b = 0; b < NB; ++b) {
#pragma unroll
      for (int fn = 0; fn < FN; ++fn) {
        v16bf bfr = *(const v16bf*)&lsB[buf][b][rB0 + fn * 16][kbB];
#pragma unroll
        for (int fm = 0; fm < FM; ++fm) {
          acc[fm][fn][b] = __builtin_amdgcn_wmma_f32_16x16x32_bf16(
              false, afr[fm], false, bfr, (short)0, acc[fm][fn][b], false, false);
        }
      }
    }
    buf ^= 1;
  }

  // Epilogue: C/D layout -> lane covers col n=lane%16, rows (lane/16)*8 + v
  const int ln = lane & 15;
  const int lm = (lane >> 4) * 8;
#pragma unroll
  for (int fm = 0; fm < FM; ++fm) {
#pragma unroll
    for (int fn = 0; fn < FN; ++fn) {
      const int gn = n0 + wn * (FN * 16) + fn * 16 + ln;
      float b0v = bias0[gn];
      float b1v = 0.f, b2v = 0.f;
      if constexpr (NB > 1) b1v = bias1[gn];
      if constexpr (NB > 2) b2v = bias2[gn];
#pragma unroll
      for (int v = 0; v < 8; ++v) {
        const int gm = m0 + wm * (FM * 16) + fm * 16 + lm + v;
        float a0 = acc[fm][fn][0][v] + b0v;
        float r;
        if constexpr (MODE == 0) {
          float a1 = acc[fm][fn][1][v] + b1v;
          float a2 = acc[fm][fn][2][v] + b2v;
          // f=sig(a0), c=silu(a1), g=sig(a2); h_prev=0 => gh = g*(1-f)*c
          r = sigm(a2) * (1.0f - sigm(a0)) * siluf(a1);
        } else if constexpr (MODE == 2) {
          float a1 = acc[fm][fn][1][v] + b1v;
          r = sigm(a0) * siluf(a1);        // gg * u
        } else {
          r = a0;
        }
        if constexpr (MODE == 3) {
          ((float*)Outp)[(size_t)gm * N + gn] = r;
        } else {
          ((__bf16*)Outp)[(size_t)gm * N + gn] = f2bf(r);
        }
      }
    }
  }
}

extern "C" void kernel_launch(void* const* d_in, const int* in_sizes, int n_in,
                              void* d_out, int out_size, void* d_ws, size_t ws_size,
                              hipStream_t stream) {
  (void)in_sizes; (void)n_in; (void)out_size; (void)ws_size;
  const int M = 4096, D = 2048, H = 8192;

  const float* x          = (const float*)d_in[0];
  const float* f_gate_w   = (const float*)d_in[1];
  const float* f_gate_b   = (const float*)d_in[2];
  const float* c_proj_w   = (const float*)d_in[3];
  const float* c_proj_b   = (const float*)d_in[4];
  const float* g_gate_w   = (const float*)d_in[5];
  const float* g_gate_b   = (const float*)d_in[6];
  const float* out_proj_w = (const float*)d_in[7];
  const float* out_proj_b = (const float*)d_in[8];
  const float* proj_u_w   = (const float*)d_in[9];
  const float* proj_u_b   = (const float*)d_in[10];
  const float* proj_g_w   = (const float*)d_in[11];
  const float* proj_g_b   = (const float*)d_in[12];
  const float* proj_out_w = (const float*)d_in[13];
  const float* proj_out_b = (const float*)d_in[14];

  // workspace: gh [M,D] bf16, o [M,D] bf16, t [M,H] bf16  (~101 MB)
  __bf16* gh = (__bf16*)d_ws;
  __bf16* o  = gh + (size_t)M * D;
  __bf16* tt = o  + (size_t)M * D;

  dim3 blk(256);

  // 1) fused f/c/g GEMMs + MLGRU gating -> gh (bf16); block tile 128x64
  bit_gemm<0><<<dim3(D / 64, M / 128), blk, 0, stream>>>(
      x, f_gate_w, c_proj_w, g_gate_w, f_gate_b, c_proj_b, g_gate_b,
      gh, M, D, D);

  // 2) o = gh @ tern(out_proj_w)^T + b  (bf16); block tile 128x128
  bit_gemm<1><<<dim3(D / 128, M / 128), blk, 0, stream>>>(
      gh, out_proj_w, nullptr, nullptr, out_proj_b, nullptr, nullptr,
      o, M, D, D);

  // 3) fused BitGLU gates: t = sig(o@Wg^T+bg) * silu(o@Wu^T+bu); block 128x128
  bit_gemm<2><<<dim3(H / 128, M / 128), blk, 0, stream>>>(
      o, proj_g_w, proj_u_w, nullptr, proj_g_b, proj_u_b, nullptr,
      tt, M, H, D);

  // 4) out = t @ tern(proj_out_w)^T + b  (f32 -> d_out); block tile 128x256
  bit_gemm<3><<<dim3(D / 256, M / 128), blk, 0, stream>>>(
      tt, proj_out_w, nullptr, nullptr, proj_out_b, nullptr, nullptr,
      d_out, M, D, H);
}
